// EnhancedDownSample_36447092474490
// MI455X (gfx1250) — matrix operand order repaired and checked by reading, not compiled, not run on hardware
//
#include <hip/hip_runtime.h>
#include <math.h>

#define B_      8
#define CIN     128
#define L_IN    8192
#define COUT    256
#define KCONV   5
#define LOUT    4096
#define NBLK    3
#define D_INNER 512
#define D_STATE 16
#define D_CONV  4
#define DT_RANK 16
#define EPSV    1e-5f
#define MROWS   (B_ * LOUT)   // 32768 sequence rows

typedef __bf16 bf16;
typedef __attribute__((ext_vector_type(16))) __bf16 v16bf;
typedef __attribute__((ext_vector_type(8)))  float  v8f;

__device__ __forceinline__ float silu_f(float x) { return x / (1.f + __expf(-x)); }

// ---------------------------------------------------------------------------
// WMMA GEMM: C(M,N) = A(M,K; lda) @ W(N,K)^T, fp32 in/out, 3-term bf16 split
// (a = a_hi + a_lo -> hi*hi + hi*lo + lo*hi with f32 accumulate).
// EPI: 0 = plain store, 1 = softplus(acc + bias[n])
// Block: 256 threads (8 wave32s), 64x64 C tile, BK=32, double-buffered LDS.
// Requires: K % 8 == 0, rows of A and W 16-byte aligned (true for all calls).
// ---------------------------------------------------------------------------
template <int EPI>
__global__ __launch_bounds__(256)
void wmma_gemm_kernel(const float* __restrict__ A, int lda,
                      const float* __restrict__ W,      // (N,K) row-major
                      const float* __restrict__ bias,
                      float* __restrict__ C, int ldc,
                      int M, int N, int K)
{
    __shared__ bf16 Ah[2][64 * 32], Al[2][64 * 32];
    __shared__ bf16 Bh[2][64 * 32], Bl[2][64 * 32];   // 32 KB total

    const int tid  = threadIdx.x;
    const int lane = tid & 31;
    const int wave = tid >> 5;
    const int mt   = wave & 3;            // m-tile 0..3
    const int nt0  = (wave >> 2) << 1;    // n-tile pair base: 0 or 2
    const int m0   = blockIdx.y * 64;
    const int n0   = blockIdx.x * 64;

    const int lrow = tid >> 2;            // tile row this thread stages (0..63)
    const int lcol = (tid & 3) * 8;       // 8 consecutive k per thread
    const int gm   = m0 + lrow;
    const int gn   = n0 + lrow;

    v8f acc0 = {0.f, 0.f, 0.f, 0.f, 0.f, 0.f, 0.f, 0.f};
    v8f acc1 = {0.f, 0.f, 0.f, 0.f, 0.f, 0.f, 0.f, 0.f};

    float4 ra0, ra1, rb0, rb1;

    // vectorized, uniformly-predicated tile fetch (2x b128 per matrix)
    auto fetch = [&](int k0) {
        const float4 z = make_float4(0.f, 0.f, 0.f, 0.f);
        ra0 = z; ra1 = z; rb0 = z; rb1 = z;
        const int gk = k0 + lcol;
        const bool kin = (gk + 8 <= K);
        if (gm < M && kin) {
            const float4* p = (const float4*)(A + (size_t)gm * lda + gk);
            ra0 = p[0]; ra1 = p[1];
        }
        if (gn < N && kin) {
            const float4* q = (const float4*)(W + (size_t)gn * K + gk);
            rb0 = q[0]; rb1 = q[1];
        }
    };

    // fp32 -> bf16 hi/lo split into LDS buffer pp
    auto stage = [&](int pp) {
        const float va[8] = {ra0.x, ra0.y, ra0.z, ra0.w, ra1.x, ra1.y, ra1.z, ra1.w};
        const float vb[8] = {rb0.x, rb0.y, rb0.z, rb0.w, rb1.x, rb1.y, rb1.z, rb1.w};
        #pragma unroll
        for (int j = 0; j < 8; ++j) {
            bf16 ha = (bf16)va[j];
            Ah[pp][lrow * 32 + lcol + j] = ha;
            Al[pp][lrow * 32 + lcol + j] = (bf16)(va[j] - (float)ha);
            bf16 hb = (bf16)vb[j];
            Bh[pp][lrow * 32 + lcol + j] = hb;
            Bl[pp][lrow * 32 + lcol + j] = (bf16)(vb[j] - (float)hb);
        }
    };

    const int nk = (K + 31) / 32;
    fetch(0);
    stage(0);
    int p = 0;

    for (int kb = 0; kb < nk; ++kb) {
        const bool more = (kb + 1 < nk);
        if (more) fetch((kb + 1) * 32);   // prefetch next tile into registers
        __syncthreads();                  // buffer p ready; buffer p^1 free

        // ---- load fragments, contiguous k = 16*(lane/16) + s in both A and B ----
        v16bf ah, al, bh0, bl0, bh1, bl1;
        {
            const int arow  = mt * 16 + (lane & 15);
            const int abase = (lane >> 4) * 16;         // K sub-chunk per half-wave
            const bf16* pAh = &Ah[p][arow * 32];
            const bf16* pAl = &Al[p][arow * 32];
            #pragma unroll
            for (int t = 0; t < 8; ++t) {
                ah[t]     = pAh[abase + t];
                ah[8 + t] = pAh[abase + 8 + t];
                al[t]     = pAl[abase + t];
                al[8 + t] = pAl[abase + 8 + t];
            }
            const int bcol = lane & 15;
            const int kb16 = (lane >> 4) * 16;
            const bf16* p0h = &Bh[p][(nt0 * 16 + bcol) * 32 + kb16];
            const bf16* p0l = &Bl[p][(nt0 * 16 + bcol) * 32 + kb16];
            const bf16* p1h = &Bh[p][((nt0 + 1) * 16 + bcol) * 32 + kb16];
            const bf16* p1l = &Bl[p][((nt0 + 1) * 16 + bcol) * 32 + kb16];
            #pragma unroll
            for (int t = 0; t < 16; ++t) {
                bh0[t] = p0h[t]; bl0[t] = p0l[t];
                bh1[t] = p1h[t]; bl1[t] = p1l[t];
            }
        }

        // ---- 3-term bf16 emulation of fp32 product, f32 accumulate ----
        acc0 = __builtin_amdgcn_wmma_f32_16x16x32_bf16(false, ah, false, bh0, (short)0, acc0, false, false);
        acc0 = __builtin_amdgcn_wmma_f32_16x16x32_bf16(false, ah, false, bl0, (short)0, acc0, false, false);
        acc0 = __builtin_amdgcn_wmma_f32_16x16x32_bf16(false, al, false, bh0, (short)0, acc0, false, false);
        acc1 = __builtin_amdgcn_wmma_f32_16x16x32_bf16(false, ah, false, bh1, (short)0, acc1, false, false);
        acc1 = __builtin_amdgcn_wmma_f32_16x16x32_bf16(false, ah, false, bl1, (short)0, acc1, false, false);
        acc1 = __builtin_amdgcn_wmma_f32_16x16x32_bf16(false, al, false, bh1, (short)0, acc1, false, false);

        if (more) stage(p ^ 1);           // fill the other buffer for next step
        p ^= 1;
    }

    // ---- epilogue: C/D layout (VGPR r: lanes 0-15 -> M=r, lanes 16-31 -> M=8+r) ----
    const int ccol  = lane & 15;
    const int rbase = (lane >> 4) * 8;
    #pragma unroll
    for (int e = 0; e < 2; ++e) {
        const int n = n0 + (nt0 + e) * 16 + ccol;
        if (n < N) {
            v8f& acc = e ? acc1 : acc0;
            #pragma unroll
            for (int r = 0; r < 8; ++r) {
                const int m = m0 + mt * 16 + rbase + r;
                if (m < M) {
                    float val = acc[r];
                    if (EPI == 1) {
                        val += bias[n];
                        val = (val > 20.f) ? val : log1pf(__expf(val));   // softplus
                    }
                    C[(size_t)m * ldc + n] = val;
                }
            }
        }
    }
}

// ---------------------------------------------------------------------------
// Front-end conv1d (stride 2, pad 2) + bias -> h (B, COUT, LOUT)
// Unguarded interior fast path; only lo==0 / lo==LOUT-1 touch the pad.
// ---------------------------------------------------------------------------
__global__ __launch_bounds__(256)
void conv1_kernel(const float* __restrict__ x, const float* __restrict__ w,
                  const float* __restrict__ bias, float* __restrict__ h)
{
    const int idx = blockIdx.x * 256 + threadIdx.x;      // B*COUT*LOUT = 2^23
    const int lo = idx & (LOUT - 1);
    const int co = (idx >> 12) & (COUT - 1);
    const int b  = idx >> 20;
    const int libase = lo * 2 - 2;
    const float* xb = x + (size_t)b * CIN * L_IN;
    const float* wc = w + (size_t)co * CIN * KCONV;
    float acc = bias[co];
    if (lo >= 1 && lo < LOUT - 1) {
        // li in [0, L_IN-2]: no bounds checks
        for (int ci = 0; ci < CIN; ++ci) {
            const float* xr = xb + (size_t)ci * L_IN + libase;
            const float* wr = wc + ci * KCONV;
            #pragma unroll
            for (int k = 0; k < KCONV; ++k) acc += xr[k] * wr[k];
        }
    } else {
        for (int ci = 0; ci < CIN; ++ci) {
            const float* xr = xb + (size_t)ci * L_IN;
            const float* wr = wc + ci * KCONV;
            #pragma unroll
            for (int k = 0; k < KCONV; ++k) {
                const int li = libase + k;
                if (li >= 0 && li < L_IN) acc += xr[li] * wr[k];
            }
        }
    }
    h[idx] = acc;
}

__global__ __launch_bounds__(256)
void bn_stats_kernel(const float* __restrict__ h, float* __restrict__ stats)
{
    const int c = blockIdx.x, tid = threadIdx.x;
    float s = 0.f, s2 = 0.f;
    for (int j = tid; j < B_ * LOUT; j += 256) {
        const int b = j >> 12, l = j & (LOUT - 1);
        const float v = h[((size_t)(b * COUT + c)) * LOUT + l];
        s += v; s2 += v * v;
    }
    __shared__ float sh[256], sh2[256];
    sh[tid] = s; sh2[tid] = s2;
    __syncthreads();
    for (int off = 128; off > 0; off >>= 1) {
        if (tid < off) { sh[tid] += sh[tid + off]; sh2[tid] += sh2[tid + off]; }
        __syncthreads();
    }
    if (tid == 0) {
        const float inv = 1.f / (B_ * LOUT);
        const float m = sh[0] * inv;
        stats[c] = m;
        stats[COUT + c] = sh2[0] * inv - m * m;
    }
}

__global__ __launch_bounds__(256)
void bn_apply_kernel(const float* __restrict__ h, const float* __restrict__ stats,
                     const float* __restrict__ g, const float* __restrict__ bb,
                     float* __restrict__ xs)
{
    const int idx = blockIdx.x * 256 + threadIdx.x;
    const int lo = idx & (LOUT - 1);
    const int co = (idx >> 12) & (COUT - 1);
    const int b  = idx >> 20;
    const float m = stats[co], v = stats[COUT + co];
    float t = (h[idx] - m) * rsqrtf(v + EPSV) * g[co] + bb[co];
    t = silu_f(t);
    xs[((size_t)(b * LOUT + lo)) * COUT + co] = t;   // transpose to (B, L, C)
}

__global__ __launch_bounds__(256)
void resid_kernel(const float* __restrict__ xs, float* __restrict__ res, int first)
{
    const int idx = blockIdx.x * 256 + threadIdx.x;
    res[idx] = first ? xs[idx] : (xs[idx] + res[idx]);
}

// wave-per-row LayerNorm over COUT=256 (8 elems/lane, wave32 shuffle reduce)
__global__ __launch_bounds__(256)
void ln_kernel(const float* __restrict__ in, const float* __restrict__ w,
               const float* __restrict__ b, float* __restrict__ out)
{
    const int row  = blockIdx.x * 8 + (threadIdx.x >> 5);
    const int lane = threadIdx.x & 31;
    const float* p = in + (size_t)row * COUT;
    float v[8], s = 0.f;
    #pragma unroll
    for (int j = 0; j < 8; ++j) { v[j] = p[lane + j * 32]; s += v[j]; }
    #pragma unroll
    for (int off = 16; off > 0; off >>= 1) s += __shfl_xor(s, off, 32);
    const float mean = s * (1.f / COUT);
    float q = 0.f;
    #pragma unroll
    for (int j = 0; j < 8; ++j) { const float d = v[j] - mean; q += d * d; }
    #pragma unroll
    for (int off = 16; off > 0; off >>= 1) q += __shfl_xor(q, off, 32);
    const float inv = rsqrtf(q * (1.f / COUT) + EPSV);
    float* o = out + (size_t)row * COUT;
    #pragma unroll
    for (int j = 0; j < 8; ++j) {
        const int c = lane + j * 32;
        o[c] = (v[j] - mean) * inv * w[c] + b[c];
    }
}

// causal depthwise conv (D_CONV=4) + bias + SiLU on u half of xz
__global__ __launch_bounds__(256)
void dwconv_kernel(const float* __restrict__ xz, const float* __restrict__ w,
                   const float* __restrict__ bias, float* __restrict__ u)
{
    const int idx = blockIdx.x * 256 + threadIdx.x;      // B*LOUT*D_INNER = 2^24
    const int d = idx & (D_INNER - 1);
    const int l = (idx >> 9) & (LOUT - 1);
    const int b = idx >> 21;
    const float* wr = w + d * D_CONV;
    float acc = bias[d];
    #pragma unroll
    for (int k = 0; k < D_CONV; ++k) {
        const int ls = l - (D_CONV - 1) + k;
        if (ls >= 0)
            acc += xz[((size_t)(b * LOUT + ls)) * (2 * D_INNER) + d] * wr[k];
    }
    u[idx] = silu_f(acc);
}

// selective scan: thread = (b, d), 16-wide state in registers, Bm/Cm via LDS.
// y = (scan + u*D) * silu(z), written in-place over dt buffer.
__global__ __launch_bounds__(256)
void scan_kernel(const float* __restrict__ u, const float* __restrict__ dt,
                 const float* __restrict__ xdbl, const float* __restrict__ xz,
                 const float* __restrict__ A_log, const float* __restrict__ Dp,
                 float* __restrict__ y)
{
    const int b = blockIdx.x >> 1;
    const int d = ((blockIdx.x & 1) << 8) + threadIdx.x;
    float As[D_STATE], hs[D_STATE];
    #pragma unroll
    for (int s = 0; s < D_STATE; ++s) {
        As[s] = -__expf(A_log[d * D_STATE + s]);
        hs[s] = 0.f;
    }
    const float Dv = Dp[d];
    __shared__ float sBC[2 * D_STATE];
    for (int l = 0; l < LOUT; ++l) {
        const size_t rowi = (size_t)(b * LOUT + l);
        if (threadIdx.x < 2 * D_STATE)
            sBC[threadIdx.x] = xdbl[rowi * 48 + DT_RANK + threadIdx.x];
        __syncthreads();
        const size_t base = rowi * D_INNER + d;
        const float dtv = dt[base];
        const float uv  = u[base];
        const float du  = dtv * uv;
        float yv = 0.f;
        #pragma unroll
        for (int s = 0; s < D_STATE; ++s) {
            hs[s] = __expf(dtv * As[s]) * hs[s] + du * sBC[s];
            yv += hs[s] * sBC[D_STATE + s];
        }
        yv += uv * Dv;
        const float z = xz[rowi * (2 * D_INNER) + D_INNER + d];
        yv *= silu_f(z);
        y[base] = yv;
        __syncthreads();
    }
}

// final: LN(x_seq + residual) -> d_out (transposed NCL) ; residual -> d_out2
__global__ __launch_bounds__(256)
void final_kernel(const float* __restrict__ xs, const float* __restrict__ res,
                  const float* __restrict__ w, const float* __restrict__ b,
                  float* __restrict__ out, float* __restrict__ outres)
{
    const int row  = blockIdx.x * 8 + (threadIdx.x >> 5);   // b*LOUT + l
    const int lane = threadIdx.x & 31;
    float v[8], r[8], s = 0.f;
    #pragma unroll
    for (int j = 0; j < 8; ++j) {
        const int c = lane + j * 32;
        r[j] = res[(size_t)row * COUT + c];
        v[j] = xs[(size_t)row * COUT + c] + r[j];
        s += v[j];
    }
    #pragma unroll
    for (int off = 16; off > 0; off >>= 1) s += __shfl_xor(s, off, 32);
    const float mean = s * (1.f / COUT);
    float q = 0.f;
    #pragma unroll
    for (int j = 0; j < 8; ++j) { const float d = v[j] - mean; q += d * d; }
    #pragma unroll
    for (int off = 16; off > 0; off >>= 1) q += __shfl_xor(q, off, 32);
    const float inv = rsqrtf(q * (1.f / COUT) + EPSV);
    const int bb = row >> 12, l = row & (LOUT - 1);
    #pragma unroll
    for (int j = 0; j < 8; ++j) {
        const int c = lane + j * 32;
        out[((size_t)(bb * COUT + c)) * LOUT + l] = (v[j] - mean) * inv * w[c] + b[c];
        outres[(size_t)row * COUT + c] = r[j];
    }
}

// ---------------------------------------------------------------------------
extern "C" void kernel_launch(void* const* d_in, const int* in_sizes, int n_in,
                              void* d_out, int out_size, void* d_ws, size_t ws_size,
                              hipStream_t stream)
{
    (void)in_sizes; (void)n_in; (void)out_size; (void)ws_size;
    const float* x        = (const float*)d_in[0];
    const float* conv_w   = (const float*)d_in[1];
    const float* conv_b   = (const float*)d_in[2];
    const float* bn_g     = (const float*)d_in[3];
    const float* bn_b     = (const float*)d_in[4];
    const float* blk_ln_w = (const float*)d_in[5];
    const float* blk_ln_b = (const float*)d_in[6];
    const float* in_proj  = (const float*)d_in[7];
    const float* convd_w  = (const float*)d_in[8];
    const float* convd_b  = (const float*)d_in[9];
    const float* xproj_w  = (const float*)d_in[10];
    const float* dtproj_w = (const float*)d_in[11];
    const float* dtproj_b = (const float*)d_in[12];
    const float* A_log    = (const float*)d_in[13];
    const float* Dp       = (const float*)d_in[14];
    const float* outproj  = (const float*)d_in[15];
    const float* lnf_w    = (const float*)d_in[16];
    const float* lnf_b    = (const float*)d_in[17];

    float* ws = (float*)d_ws;
    size_t off = 0;
    float* h     = ws + off; off += (size_t)B_ * COUT * LOUT;        // reused as xn
    float* xseq  = ws + off; off += (size_t)MROWS * COUT;
    float* resid = ws + off; off += (size_t)MROWS * COUT;
    float* xz    = ws + off; off += (size_t)MROWS * 2 * D_INNER;
    float* ubuf  = ws + off; off += (size_t)MROWS * D_INNER;
    float* dtbuf = ws + off; off += (size_t)MROWS * D_INNER;         // y in-place
    float* xdbl  = ws + off; off += (size_t)MROWS * 48;
    float* stats = ws + off; off += 2 * COUT;
    float* xn = h;                                                    // h dead after bn_apply

    conv1_kernel<<<(B_ * COUT * LOUT) / 256, 256, 0, stream>>>(x, conv_w, conv_b, h);
    bn_stats_kernel<<<COUT, 256, 0, stream>>>(h, stats);
    bn_apply_kernel<<<(B_ * COUT * LOUT) / 256, 256, 0, stream>>>(h, stats, bn_g, bn_b, xseq);

    for (int i = 0; i < NBLK; ++i) {
        resid_kernel<<<(MROWS * COUT) / 256, 256, 0, stream>>>(xseq, resid, i == 0);
        ln_kernel<<<MROWS / 8, 256, 0, stream>>>(resid, blk_ln_w + i * COUT, blk_ln_b + i * COUT, xn);

        // xz = xn @ in_proj[i]^T : M=32768, N=1024, K=256
        dim3 g1((2 * D_INNER) / 64, MROWS / 64);
        wmma_gemm_kernel<0><<<g1, 256, 0, stream>>>(
            xn, COUT, in_proj + (size_t)i * 2 * D_INNER * COUT, nullptr,
            xz, 2 * D_INNER, MROWS, 2 * D_INNER, COUT);

        dwconv_kernel<<<(MROWS * D_INNER) / 256, 256, 0, stream>>>(
            xz, convd_w + (size_t)i * D_INNER * D_CONV, convd_b + i * D_INNER, ubuf);

        // x_dbl = u @ xproj[i]^T : M=32768, N=48, K=512
        dim3 g2((48 + 63) / 64, MROWS / 64);
        wmma_gemm_kernel<0><<<g2, 256, 0, stream>>>(
            ubuf, D_INNER, xproj_w + (size_t)i * 48 * D_INNER, nullptr,
            xdbl, 48, MROWS, 48, D_INNER);

        // dt = softplus(x_dbl[:, :16] @ dtproj[i]^T + b) : N=512, K=16 (strided A)
        dim3 g3(D_INNER / 64, MROWS / 64);
        wmma_gemm_kernel<1><<<g3, 256, 0, stream>>>(
            xdbl, 48, dtproj_w + (size_t)i * D_INNER * DT_RANK, dtproj_b + i * D_INNER,
            dtbuf, D_INNER, MROWS, D_INNER, DT_RANK);

        scan_kernel<<<B_ * 2, 256, 0, stream>>>(
            ubuf, dtbuf, xdbl, xz,
            A_log + (size_t)i * D_INNER * D_STATE, Dp + i * D_INNER, dtbuf);

        // x_seq = y @ outproj[i]^T : N=256, K=512
        dim3 g4(COUT / 64, MROWS / 64);
        wmma_gemm_kernel<0><<<g4, 256, 0, stream>>>(
            dtbuf, D_INNER, outproj + (size_t)i * COUT * D_INNER, nullptr,
            xseq, COUT, MROWS, COUT, D_INNER);
    }

    final_kernel<<<MROWS / 8, 256, 0, stream>>>(
        xseq, resid, lnf_w, lnf_b,
        (float*)d_out, (float*)d_out + (size_t)B_ * COUT * LOUT);
}
